// Spectral_Atten_4380866642492
// MI455X (gfx1250) — matrix operand order repaired and checked by reading, not compiled, not run on hardware
//
#include <hip/hip_runtime.h>
#include <hip/hip_bf16.h>

// ---------------------------------------------------------------------------
// Spectral attention for MI455X (gfx1250, wave32, WMMA bf16 16x16x32 f32-acc)
// ---------------------------------------------------------------------------

#define BATCH 4
#define CDIM 128
#define HDIM 256
#define WDIM 256
#define NPIX (HDIM * WDIM)   // 65536
#define NHEADS 8

// Transposed-tile ci stride (shorts): multiple of 8 for b128 alignment.
#define CSTR 136

typedef __attribute__((ext_vector_type(16))) __bf16    v16bf;
typedef __attribute__((ext_vector_type(8)))  float     v8f;
typedef __attribute__((ext_vector_type(4)))  float     v4f;
typedef __attribute__((ext_vector_type(4)))  unsigned  v4u;

// gcc-style int vector: matches the clang builtin prototype for async LDS copies
typedef int v4i __attribute__((vector_size(16)));

union Frag { v4u q[2]; v16bf v; };   // 32 bytes: one WMMA A/B operand

#if defined(__has_builtin)
#  if __has_builtin(__builtin_amdgcn_global_load_async_to_lds_b128)
#    define HAVE_ASYNC_LDS 1
#  endif
#  if __has_builtin(__builtin_amdgcn_s_wait_asynccnt)
#    define HAVE_WAIT_ASYNC 1
#  endif
#endif
#ifndef HAVE_ASYNC_LDS
#  define HAVE_ASYNC_LDS 0
#endif
#ifndef HAVE_WAIT_ASYNC
#  define HAVE_WAIT_ASYNC 0
#endif

#if HAVE_ASYNC_LDS
typedef __attribute__((address_space(1))) v4i ga_v4i;
typedef __attribute__((address_space(3))) v4i ls_v4i;
#endif

__device__ __forceinline__ unsigned short f2bf(float f) {
  union { float f; unsigned u; } v; v.f = f;
  unsigned r = (v.u + 0x7FFFu + ((v.u >> 16) & 1u)) >> 16;  // RNE
  return (unsigned short)r;
}
__device__ __forceinline__ float bf2f(unsigned short s) {
  union { unsigned u; float f; } v; v.u = ((unsigned)s) << 16; return v.f;
}
__device__ __forceinline__ __bf16 bfbits(unsigned short s) {
  union { unsigned short u; __bf16 b; } v; v.u = s; return v.b;
}

// A-matrix 16x32 bf16 element->K map (ISA 7.12.2): lane half selects K block.
__device__ __forceinline__ int amapK(int e, int hi) {
  return (e < 8) ? (hi * 8 + e) : (16 + hi * 8 + (e - 8));
}

// Pack one A fragment (row `Wrow`, K chunk kk*32) from f32 weights.
// Per-lane data = two contiguous 8-float runs -> 4x global_load_b128.
__device__ __forceinline__ v16bf packA(const float* __restrict__ Wrow, int kk, int hi) {
  const float* p0 = Wrow + kk * 32 + hi * 8;
  const float* p1 = Wrow + kk * 32 + 16 + hi * 8;
  v4f w0 = *(const v4f*)(p0);
  v4f w1 = *(const v4f*)(p0 + 4);
  v4f w2 = *(const v4f*)(p1);
  v4f w3 = *(const v4f*)(p1 + 4);
  v16bf a;
  #pragma unroll
  for (int e = 0; e < 4; ++e) {
    a[e]      = bfbits(f2bf(w0[e]));
    a[e + 4]  = bfbits(f2bf(w1[e]));
    a[e + 8]  = bfbits(f2bf(w2[e]));
    a[e + 12] = bfbits(f2bf(w3[e]));
  }
  return a;
}

// B fragment from a transposed LDS tile: two 8-bf16 runs (K table order) -> 2x ds_load_b128.
__device__ __forceinline__ v16bf loadB(const unsigned short* __restrict__ tile,
                                       int p, int kk, int hi) {
  Frag f;
  const unsigned short* s = tile + p * CSTR + kk * 32 + hi * 8;
  f.q[0] = *(const v4u*)(s);
  f.q[1] = *(const v4u*)(s + 16);
  return f.v;
}

// ---------------------------------------------------------------------------
// Kernel 1: q/k/v = dwconv3x3(conv1x1(x, W)); also accumulate sum(q^2), sum(k^2)
// Tile: 16x4 output pixels, halo 18x6 = 108 pixels (padded to 112).
// ---------------------------------------------------------------------------
__global__ __launch_bounds__(256) void k_qkv(
    const float* __restrict__ x,
    const float* __restrict__ Wq, const float* __restrict__ Wk, const float* __restrict__ Wv,
    const float* __restrict__ dwq, const float* __restrict__ dwk, const float* __restrict__ dwv,
    unsigned short* __restrict__ qo, unsigned short* __restrict__ ko, unsigned short* __restrict__ vo,
    float* __restrict__ qsq, float* __restrict__ ksq)
{
  __shared__ __align__(16) unsigned short xs[112 * CSTR];  // transposed [p][ci]
  __shared__ __align__(16) unsigned short ts[CDIM * 112];  // [co][p]
  __shared__ float dwc[CDIM * 9];
  __shared__ float red[CDIM];

  const int tid  = threadIdx.x;
  const int wave = tid >> 5, lane = tid & 31;
  const int m = lane & 15, hi = lane >> 4;
  const int tx = blockIdx.x * 16, ty = blockIdx.y * 4, b = blockIdx.z;

  for (int i = tid; i < CDIM * 112; i += 256) {
    int ch = i / 112, p = i % 112;
    float v = 0.f;
    if (p < 108) {
      int py = p / 18, px = p % 18;
      int gy = ty + py - 1, gx = tx + px - 1;
      if (gy >= 0 && gy < HDIM && gx >= 0 && gx < WDIM)
        v = x[(((size_t)b * CDIM + ch) * HDIM + gy) * WDIM + gx];
    }
    xs[p * CSTR + ch] = f2bf(v);
  }
  if (tid < CDIM) red[tid] = 0.f;
  __syncthreads();

  const float* Ws[3] = {Wq, Wk, Wv};
  const float* Ds[3] = {dwq, dwk, dwv};
  unsigned short* Os[3] = {qo, ko, vo};
  float* Sq[2] = {qsq, ksq};

  for (int s = 0; s < 3; ++s) {
    for (int i = tid; i < CDIM * 9; i += 256) dwc[i] = Ds[s][i];

    // conv1x1 GEMM: ts[co][p] = sum_ci W[co][ci] * xs[p][ci]
    const float* Wrow = Ws[s] + (wave * 16 + m) * CDIM;
    const int co_base = wave * 16;  // 8 waves x 16 rows = 128 co
    v16bf A[4];
    #pragma unroll
    for (int kk = 0; kk < 4; ++kk) A[kk] = packA(Wrow, kk, hi);

    for (int pt = 0; pt < 7; ++pt) {          // 7*16 = 112 pixels (pad rows = 0)
      v8f acc = {};
      #pragma unroll
      for (int kk = 0; kk < 4; ++kk) {
        v16bf Bf = loadB(xs, pt * 16 + m, kk, hi);
        acc = __builtin_amdgcn_wmma_f32_16x16x32_bf16(false, A[kk], false, Bf,
                                                      (short)0, acc, false, false);
      }
      #pragma unroll
      for (int r = 0; r < 8; ++r)
        ts[(co_base + hi * 8 + r) * 112 + pt * 16 + m] = f2bf(acc[r]);
    }
    __syncthreads();

    // depthwise 3x3 on ts, store bf16, accumulate sum of squares (q,k only)
    for (int i = tid; i < CDIM * 64; i += 256) {
      int ch = i >> 6, p = i & 63;
      int py = p >> 4, px = p & 15;
      float sum = 0.f;
      #pragma unroll
      for (int ky = 0; ky < 3; ++ky)
        #pragma unroll
        for (int kx = 0; kx < 3; ++kx)
          sum += dwc[ch * 9 + ky * 3 + kx] * bf2f(ts[ch * 112 + (py + ky) * 18 + (px + kx)]);
      Os[s][((size_t)b * CDIM + ch) * NPIX + (ty + py) * WDIM + (tx + px)] = f2bf(sum);
      if (s < 2) atomicAdd(&red[ch], sum * sum);
    }
    __syncthreads();
    if (s < 2) {
      if (tid < CDIM) { atomicAdd(&Sq[s][b * CDIM + tid], red[tid]); red[tid] = 0.f; }
      __syncthreads();
    }
  }
}

// ---------------------------------------------------------------------------
// Kernel 2: msum[b][co] += sum over tile pixels of relu(conv3x3(mask, Wm1))
// K reordered as (tap r9 outer, ci inner) so im2col B-fragments are contiguous.
// ---------------------------------------------------------------------------
__global__ __launch_bounds__(256) void k_mask(
    const float* __restrict__ mask, const float* __restrict__ Wm1,
    float* __restrict__ msum)
{
  __shared__ __align__(16) unsigned short ms[108 * CSTR];  // transposed [p][ci]
  __shared__ float red[CDIM];

  const int tid  = threadIdx.x;
  const int wave = tid >> 5, lane = tid & 31;
  const int m = lane & 15, hi = lane >> 4;
  const int tx = blockIdx.x * 16, ty = blockIdx.y * 4, b = blockIdx.z;

  for (int i = tid; i < CDIM * 108; i += 256) {
    int ch = i / 108, p = i % 108;
    int py = p / 18, px = p % 18;
    int gy = ty + py - 1, gx = tx + px - 1;
    float v = 0.f;
    if (gy >= 0 && gy < HDIM && gx >= 0 && gx < WDIM)
      v = mask[(((size_t)b * CDIM + ch) * HDIM + gy) * WDIM + gx];
    ms[p * CSTR + ch] = f2bf(v);
  }
  if (tid < CDIM) red[tid] = 0.f;
  __syncthreads();

  const int co_base = wave * 16;
  v8f a0 = {}, a1 = {}, a2 = {}, a3 = {};
  for (int r9 = 0; r9 < 9; ++r9) {
    const int ky = r9 / 3, kx = r9 % 3;
    #pragma unroll
    for (int kk = 0; kk < 4; ++kk) {
      v16bf A;
      #pragma unroll
      for (int e = 0; e < 16; ++e) {
        int ci = kk * 32 + amapK(e, hi);
        A[e] = bfbits(f2bf(Wm1[((co_base + m) * CDIM + ci) * 9 + r9]));
      }
      // output pixel p = pt*16 + m  =>  py = pt, px = m
      v16bf B0 = loadB(ms, (0 + ky) * 18 + (m + kx), kk, hi);
      a0 = __builtin_amdgcn_wmma_f32_16x16x32_bf16(false, A, false, B0, (short)0, a0, false, false);
      v16bf B1 = loadB(ms, (1 + ky) * 18 + (m + kx), kk, hi);
      a1 = __builtin_amdgcn_wmma_f32_16x16x32_bf16(false, A, false, B1, (short)0, a1, false, false);
      v16bf B2 = loadB(ms, (2 + ky) * 18 + (m + kx), kk, hi);
      a2 = __builtin_amdgcn_wmma_f32_16x16x32_bf16(false, A, false, B2, (short)0, a2, false, false);
      v16bf B3 = loadB(ms, (3 + ky) * 18 + (m + kx), kk, hi);
      a3 = __builtin_amdgcn_wmma_f32_16x16x32_bf16(false, A, false, B3, (short)0, a3, false, false);
    }
  }
  // relu + reduce over the 64 tile pixels per output channel
  #pragma unroll
  for (int r = 0; r < 8; ++r) {
    int co = co_base + hi * 8 + r;
    float s = fmaxf(a0[r], 0.f) + fmaxf(a1[r], 0.f) + fmaxf(a2[r], 0.f) + fmaxf(a3[r], 0.f);
    atomicAdd(&red[co], s);
  }
  __syncthreads();
  if (tid < CDIM) atomicAdd(&msum[b * CDIM + tid], red[tid]);
}

// ---------------------------------------------------------------------------
// Kernel 3: G[b,h,16,16] = q_bh @ k_bh^T (dot over N=65536), unnormalized.
// A/B fragments are contiguous bf16 runs in global memory -> b128 loads.
// ---------------------------------------------------------------------------
__global__ __launch_bounds__(256) void k_gram(
    const unsigned short* __restrict__ q, const unsigned short* __restrict__ k,
    float* __restrict__ G)
{
  __shared__ float Gs[256];
  const int tid  = threadIdx.x;
  const int wave = tid >> 5, lane = tid & 31;
  const int m = lane & 15, hi = lane >> 4;
  const int slab = blockIdx.x, head = blockIdx.y, b = blockIdx.z;
  const size_t base = ((size_t)b * CDIM + head * 16) * NPIX + (size_t)m * NPIX;
  const int n0 = slab * 8192 + wave * 1024;

  v8f acc = {};
  for (int kk = 0; kk < 32; ++kk) {
    const int nb = n0 + kk * 32;
    Frag A, Bf;
    const unsigned short* qrow = q + base + nb;
    const unsigned short* krow = k + base + nb;
    A.q[0]  = *(const v4u*)(qrow + hi * 8);
    A.q[1]  = *(const v4u*)(qrow + 16 + hi * 8);
    Bf.q[0] = *(const v4u*)(krow + hi * 8);
    Bf.q[1] = *(const v4u*)(krow + 16 + hi * 8);
    acc = __builtin_amdgcn_wmma_f32_16x16x32_bf16(false, A.v, false, Bf.v,
                                                  (short)0, acc, false, false);
  }
  Gs[tid] = 0.f;
  __syncthreads();
  #pragma unroll
  for (int r = 0; r < 8; ++r)
    atomicAdd(&Gs[(hi * 8 + r) * 16 + m], acc[r]);
  __syncthreads();
  atomicAdd(&G[((size_t)b * NHEADS + head) * 256 + tid], Gs[tid]);
}

// ---------------------------------------------------------------------------
// Kernel 4: bias from mask branch, normalize gram, softmax, and fold
// Wproj into the per-head attention: M_b = Wproj @ blockdiag(atten_b)  (bf16)
// ---------------------------------------------------------------------------
__global__ __launch_bounds__(256) void k_attn(
    const float* __restrict__ Wm2, const float* __restrict__ Wproj,
    const float* __restrict__ rescale,
    const float* __restrict__ msum, const float* __restrict__ qsq, const float* __restrict__ ksq,
    const float* __restrict__ G, unsigned short* __restrict__ M)
{
  const int b = blockIdx.x, tid = threadIdx.x;
  __shared__ float mbs[CDIM], qn[CDIM], kn[CDIM];
  __shared__ float att[CDIM * 16];

  if (tid < CDIM) {
    float s = 0.f;
    for (int i = 0; i < CDIM; ++i) s += Wm2[tid * CDIM + i] * msum[b * CDIM + i];
    mbs[tid] = s * (1.f / (float)NPIX);
    qn[tid] = fmaxf(sqrtf(qsq[b * CDIM + tid]), 1e-12f);
    kn[tid] = fmaxf(sqrtf(ksq[b * CDIM + tid]), 1e-12f);
  }
  __syncthreads();

  if (tid < CDIM) {                      // one softmax row per thread
    int h = tid >> 4, mm = tid & 15;
    float rs = rescale[h];
    float row[16], mx = -1e30f;
    for (int d = 0; d < 16; ++d) {
      float g = G[((size_t)b * NHEADS + h) * 256 + mm * 16 + d];
      float v = g / (qn[h * 16 + mm] * kn[h * 16 + d]) * rs
                + mbs[h * 16 + d] - mbs[h * 16 + mm];
      row[d] = v; mx = fmaxf(mx, v);
    }
    float se = 0.f;
    for (int d = 0; d < 16; ++d) { row[d] = __expf(row[d] - mx); se += row[d]; }
    float inv = 1.f / se;
    for (int d = 0; d < 16; ++d) att[tid * 16 + d] = row[d] * inv;
  }
  __syncthreads();

  for (int idx = tid; idx < CDIM * CDIM; idx += 256) {
    int o = idx >> 7, cp = idx & 127;
    int h = cp >> 4, d = cp & 15;
    float s = 0.f;
    #pragma unroll
    for (int mm = 0; mm < 16; ++mm)
      s += Wproj[o * CDIM + h * 16 + mm] * att[(h * 16 + mm) * 16 + d];
    M[((size_t)b * CDIM + o) * CDIM + cp] = f2bf(s);
  }
}

// ---------------------------------------------------------------------------
// Kernel 5: out = M_b @ v  (fused atten@v + Wproj conv1x1), f32 output.
// M_b staged to LDS via gfx1250 async global->LDS when available; v tile
// transposed in LDS so B fragments are 2x ds_load_b128.
// ---------------------------------------------------------------------------
__global__ __launch_bounds__(256) void k_out(
    const unsigned short* __restrict__ v, const unsigned short* __restrict__ M,
    float* __restrict__ out)
{
  __shared__ __align__(16) unsigned short vs[64 * CSTR];     // transposed [p][ci]
  __shared__ __align__(16) unsigned short Ms[CDIM * CDIM];   // row-major [o][c']
  const int tid  = threadIdx.x;
  const int wave = tid >> 5, lane = tid & 31;
  const int m = lane & 15, hi = lane >> 4;
  const int b = blockIdx.z;
  const int n0 = blockIdx.x * 64;

  if (n0 + 64 < NPIX)
    __builtin_prefetch(v + ((size_t)b * CDIM) * NPIX + n0 + 64, 0, 1);

  // Stage M_b (128x128 bf16 = 32 KB) into LDS.
#if HAVE_ASYNC_LDS
  {
    const unsigned short* Mg = M + (size_t)b * CDIM * CDIM;
    for (int i = tid; i < 2048; i += 256)
      __builtin_amdgcn_global_load_async_to_lds_b128(
          (ga_v4i*)(unsigned long long)(Mg + i * 8),
          (ls_v4i*)(unsigned long long)(&Ms[i * 8]), 0, 0);
  }
#else
  for (int i = tid; i < 2048; i += 256)
    *(v4u*)&Ms[i * 8] = *(const v4u*)(M + (size_t)b * CDIM * CDIM + i * 8);
#endif

  // Stage v tile transposed: vector global reads, 16-bit LDS scatter.
  for (int j = tid; j < 1024; j += 256) {          // 128 ch x 8 chunks of 8 px
    int ch = j >> 3, ck = j & 7;
    v4u d = *(const v4u*)(v + ((size_t)b * CDIM + ch) * NPIX + n0 + ck * 8);
    #pragma unroll
    for (int w = 0; w < 4; ++w) {
      vs[(ck * 8 + 2 * w) * CSTR + ch]     = (unsigned short)(d[w] & 0xffffu);
      vs[(ck * 8 + 2 * w + 1) * CSTR + ch] = (unsigned short)(d[w] >> 16);
    }
  }
#if HAVE_WAIT_ASYNC
  __builtin_amdgcn_s_wait_asynccnt(0);
#endif
  __syncthreads();

  const int co_base = wave * 16;
  Frag A[4];
  #pragma unroll
  for (int kk = 0; kk < 4; ++kk) {
    const unsigned short* row = &Ms[(co_base + m) * CDIM + kk * 32];
    A[kk].q[0] = *(const v4u*)(row + hi * 8);
    A[kk].q[1] = *(const v4u*)(row + 16 + hi * 8);
  }

  for (int pt = 0; pt < 4; ++pt) {
    v8f acc = {};
    #pragma unroll
    for (int kk = 0; kk < 4; ++kk) {
      v16bf Bf = loadB(vs, pt * 16 + m, kk, hi);
      acc = __builtin_amdgcn_wmma_f32_16x16x32_bf16(false, A[kk].v, false, Bf,
                                                    (short)0, acc, false, false);
    }
    #pragma unroll
    for (int r = 0; r < 8; ++r)
      out[((size_t)b * CDIM + co_base + hi * 8 + r) * NPIX + n0 + pt * 16 + m] = acc[r];
  }
}

// ---------------------------------------------------------------------------
extern "C" void kernel_launch(void* const* d_in, const int* in_sizes, int n_in,
                              void* d_out, int out_size, void* d_ws, size_t ws_size,
                              hipStream_t stream) {
  const float* x       = (const float*)d_in[0];
  const float* mask    = (const float*)d_in[1];
  const float* Wq      = (const float*)d_in[2];
  const float* Wk      = (const float*)d_in[3];
  const float* Wv      = (const float*)d_in[4];
  const float* dwq     = (const float*)d_in[5];
  const float* dwk     = (const float*)d_in[6];
  const float* dwv     = (const float*)d_in[7];
  const float* rescale = (const float*)d_in[8];
  const float* Wm1     = (const float*)d_in[9];
  const float* Wm2     = (const float*)d_in[10];
  const float* Wproj   = (const float*)d_in[11];

  char* ws = (char*)d_ws;
  const size_t SZ_T = (size_t)BATCH * CDIM * NPIX * sizeof(unsigned short); // 64 MB
  unsigned short* qb = (unsigned short*)(ws);
  unsigned short* kb = (unsigned short*)(ws + SZ_T);
  unsigned short* vb = (unsigned short*)(ws + 2 * SZ_T);
  float* qsq  = (float*)(ws + 3 * SZ_T);
  float* ksq  = qsq + BATCH * CDIM;
  float* msum = ksq + BATCH * CDIM;
  float* G    = msum + BATCH * CDIM;                         // 4*8*256 floats
  unsigned short* M = (unsigned short*)(G + BATCH * NHEADS * 256);

  // zero the atomic accumulators (qsq, ksq, msum, G)
  (void)hipMemsetAsync(qsq, 0, (3 * BATCH * CDIM + BATCH * NHEADS * 256) * sizeof(float), stream);

  k_qkv <<<dim3(WDIM / 16, HDIM / 4, BATCH), 256, 0, stream>>>(
      x, Wq, Wk, Wv, dwq, dwk, dwv, qb, kb, vb, qsq, ksq);
  k_mask<<<dim3(WDIM / 16, HDIM / 4, BATCH), 256, 0, stream>>>(mask, Wm1, msum);
  k_gram<<<dim3(8, NHEADS, BATCH), 256, 0, stream>>>(qb, kb, G);
  k_attn<<<dim3(BATCH), 256, 0, stream>>>(Wm2, Wproj, rescale, msum, qsq, ksq, G, M);
  k_out <<<dim3(NPIX / 64, 1, BATCH), 256, 0, stream>>>(vb, M, (float*)d_out);
}
